// MultiQueryAttention_68066641707177
// MI455X (gfx1250) — matrix operand order repaired and checked by reading, compiled, not run on hardware
//
#include <hip/hip_runtime.h>
#include <hip/hip_bf16.h>

// MQA forward for MI455X (gfx1250), wave32 + WMMA bf16 + TDM async path.
// B=2, S=2048, D=2048, H=16, HKV=2, HD=128.

#define B_     2
#define SEQ    2048
#define DMODEL 2048
#define NH     16
#define NKV    2
#define HDIM   128

typedef __bf16 bf16_t;
typedef __attribute__((ext_vector_type(16))) __bf16 v16bf;
typedef __attribute__((ext_vector_type(8)))  __bf16 v8bf;
typedef __attribute__((ext_vector_type(8)))  float  v8f;
typedef __attribute__((ext_vector_type(4)))  unsigned int v4u;
typedef __attribute__((ext_vector_type(8)))  int v8i;
typedef __attribute__((ext_vector_type(4)))  int v4i;

static __device__ __forceinline__ v8f zero_v8f() {
  v8f z = {0.f, 0.f, 0.f, 0.f, 0.f, 0.f, 0.f, 0.f};
  return z;
}

// D = A(16x32 bf16) * B(32x16 bf16) + C(16x16 f32)
static __device__ __forceinline__ v8f wmma_bf16(v16bf a, v16bf b, v8f c) {
  return __builtin_amdgcn_wmma_f32_16x16x32_bf16(
      /*neg_a=*/false, a, /*neg_b=*/false, b,
      /*c_mod=*/(short)0, c, /*reuse_a=*/false, /*reuse_b=*/false);
}

// A-fragment (16-bit A 16x32): lane group khalf holds
// K = {khalf*16 .. khalf*16+15} (contiguous, matches B fragments).
static __device__ __forceinline__ v16bf load_afrag(const bf16_t* row32, int khalf) {
  return *(const v16bf*)(row32 + khalf * 16);
}

// Generic pointer to an LDS object carries the LDS byte offset in addr[31:0]
// (flat aperture decoding, ISA 10.2).
static __device__ __forceinline__ unsigned lds_off(const void* p) {
  return (unsigned)(unsigned long long)p;
}

// Issue a TDM 2D tile load (bf16 elements): tile rows x row_elems from a
// row-major tensor (row stride == row_elems) into LDS at lds_byte_off.
// Descriptor per ISA 8.3/8.4: group0 {count=1, lds_addr, global_addr, type=2},
// group1 {data_size=2B, tensor_dim0/1, tile_dim0/1, dim0 stride}.
// Tracked by TENSORcnt. EXEC ignored -> caller must guard to one wave.
static __device__ __forceinline__ void tdm_load_2d_bf16(
    const bf16_t* gptr, unsigned lds_byte_off,
    unsigned rows, unsigned row_elems, unsigned tensor_rows)
{
  unsigned long long g = (unsigned long long)gptr;
  v4u g0;
  g0[0] = 1u;                                              // count=1, load
  g0[1] = lds_byte_off;                                    // lds_addr
  g0[2] = (unsigned)(g & 0xFFFFFFFFu);                     // global_addr lo
  g0[3] = (unsigned)((g >> 32) & 0x1FFFFFFu) | (2u << 30); // addr hi | type=2

  v8i g1;
  g1[0] = (int)(1u << 16);                                  // data_size = 2B
  g1[1] = (int)((row_elems & 0xFFFFu) << 16);               // tensor_dim0 lo
  g1[2] = (int)((row_elems >> 16) | ((tensor_rows & 0xFFFFu) << 16));
  g1[3] = (int)((tensor_rows >> 16) | ((row_elems & 0xFFFFu) << 16)); // tile_dim0
  g1[4] = (int)(rows & 0xFFFFu);                            // tile_dim1, tile_dim2=0
  g1[5] = (int)row_elems;                                   // dim0 stride lo
  g1[6] = 0;
  g1[7] = 0;

  v4i z4 = {0, 0, 0, 0};
  v8i z8 = {0, 0, 0, 0, 0, 0, 0, 0};
  // clang-23 / therock-10.0 form: (g0, g1, g2, g3, g4, cpol)
  __builtin_amdgcn_tensor_load_to_lds(g0, g1, z4, z4, z8, 0);
}

// ---------------------------------------------------------------------------
// GEMM: C[M,N] = A[M,K] * Bw[K,N].  A is f32 (A_BF16=0) or bf16 (A_BF16=1),
// Bw is f32, converted to bf16 into LDS.  Accumulate f32 via WMMA.
// epi==0: write f32 row-major C[m*N+n]
// epi==1: write bf16 into [b, hx, s, hd] with m=b*Sdim+s, n=hx*HDIM+hd
// Block: 256 threads = 8 waves, tile 128x128, K-step 32, wave tile 32x64.
// ---------------------------------------------------------------------------
template <int A_BF16>
__global__ __launch_bounds__(256)
void gemm_wmma_kernel(const void* __restrict__ Aop,
                      const float* __restrict__ Bw,
                      void* __restrict__ Cout,
                      int M, int N, int K,
                      int epi, int Sdim, int Hx)
{
  __shared__ alignas(64) bf16_t ldsA[128 * 32];   // [m][k]
  __shared__ alignas(64) bf16_t ldsBt[128 * 32];  // [n][k] (transposed)

  const int bm    = blockIdx.y * 128;
  const int bn    = blockIdx.x * 128;
  const int tid   = threadIdx.x;
  const int lane  = tid & 31;
  const int wid   = tid >> 5;
  const int wm    = (wid >> 1) * 32;  // wave row offset 0/32/64/96
  const int wn    = (wid & 1) * 64;   // wave col offset 0/64
  const int col   = lane & 15;
  const int khalf = lane >> 4;

  const float*  Af = (const float*)Aop;
  const bf16_t* Ab = (const bf16_t*)Aop;

  v8f acc[2][4];
#pragma unroll
  for (int mi = 0; mi < 2; ++mi)
#pragma unroll
    for (int ni = 0; ni < 4; ++ni)
      acc[mi][ni] = zero_v8f();

  for (int k0 = 0; k0 < K; k0 += 32) {
    if (k0 + 32 < K) {
      __builtin_prefetch((const char*)Bw + ((size_t)(k0 + 32) * N + bn) * 4, 0, 0);
    }
    __syncthreads();
    // A tile 128x32 -> ldsA (convert to bf16). 4096 elems, 16/thread, 64B runs.
#pragma unroll
    for (int i = 0; i < 16; ++i) {
      int l = tid * 16 + i;
      int m = l >> 5, k = l & 31;
      float a = A_BF16 ? (float)Ab[(size_t)(bm + m) * K + k0 + k]
                       : Af[(size_t)(bm + m) * K + k0 + k];
      ldsA[m * 32 + k] = (bf16_t)a;
    }
    // B tile 32x128 -> ldsBt transposed [n][k]
#pragma unroll
    for (int i = 0; i < 16; ++i) {
      int l = tid * 16 + i;
      int k = l >> 7, n = l & 127;
      ldsBt[n * 32 + k] = (bf16_t)Bw[(size_t)(k0 + k) * N + bn + n];
    }
    __syncthreads();

#pragma unroll
    for (int mi = 0; mi < 2; ++mi) {
      v16bf af = load_afrag(&ldsA[(wm + mi * 16 + col) * 32], khalf);
#pragma unroll
      for (int ni = 0; ni < 4; ++ni) {
        v16bf bf = *(const v16bf*)&ldsBt[(wn + ni * 16 + col) * 32 + khalf * 16];
        acc[mi][ni] = wmma_bf16(af, bf, acc[mi][ni]);
      }
    }
  }

  // Epilogue. C layout: vgpr j holds row (j + 8*khalf), lane col = N column.
#pragma unroll
  for (int mi = 0; mi < 2; ++mi)
#pragma unroll
    for (int ni = 0; ni < 4; ++ni)
#pragma unroll
      for (int j = 0; j < 8; ++j) {
        int mg = bm + wm + mi * 16 + j + 8 * khalf;
        int ng = bn + wn + ni * 16 + col;
        float v = acc[mi][ni][j];
        if (epi == 0) {
          ((float*)Cout)[(size_t)mg * N + ng] = v;
        } else {
          int b = mg / Sdim, s = mg % Sdim;
          int hx = ng / HDIM, hd = ng % HDIM;
          ((bf16_t*)Cout)[(((size_t)b * Hx + hx) * Sdim + s) * HDIM + hd] = (bf16_t)v;
        }
      }
}

// ---------------------------------------------------------------------------
// In-place RoPE on bf16 tensor laid out [B, Hx, S, HDIM].
// ---------------------------------------------------------------------------
__global__ void rope_kernel(bf16_t* __restrict__ t, int Hx)
{
  const int half = HDIM / 2;
  size_t total = (size_t)B_ * Hx * SEQ * half;
  size_t idx = (size_t)blockIdx.x * blockDim.x + threadIdx.x;
  if (idx >= total) return;
  int i = (int)(idx % half);
  int s = (int)((idx / half) % SEQ);
  size_t bh = idx / ((size_t)half * SEQ);
  bf16_t* p = t + (bh * SEQ + (size_t)s) * HDIM;

  float inv_freq = exp2f(-(2.0f * (float)i / (float)HDIM) * log2f(10000.0f));
  float ang = (float)s * inv_freq;
  float sn, c;
  __sincosf(ang, &sn, &c);
  float x1 = (float)p[i];
  float x2 = (float)p[i + half];
  p[i]        = (bf16_t)(x1 * c - x2 * sn);
  p[i + half] = (bf16_t)(x2 * c + x1 * sn);
}

// ---------------------------------------------------------------------------
// Flash attention (causal, GQA).  Block = 128 threads = 4 waves; each wave
// owns 16 q rows.  K tiles: TDM double-buffered (DMA overlaps compute);
// V tiles: vector loads + LDS transpose (overlaps in-flight TDM).
// ---------------------------------------------------------------------------
__global__ __launch_bounds__(128)
void attn_kernel(const bf16_t* __restrict__ qbuf,
                 const bf16_t* __restrict__ kbuf,
                 const bf16_t* __restrict__ vbuf,
                 bf16_t* __restrict__ abuf)
{
  const int nqt = SEQ / 64;
  const int qt  = blockIdx.x % nqt;
  const int h   = (blockIdx.x / nqt) % NH;
  const int b   = blockIdx.x / (nqt * NH);
  const int hkv = h / (NH / NKV);

  const int tid   = threadIdx.x;
  const int lane  = tid & 31;
  const int wid   = tid >> 5;
  const int col   = lane & 15;
  const int khalf = lane >> 4;
  const int qbase_wg = qt * 64;
  const int qbase    = qbase_wg + wid * 16;

  const bf16_t* Q  = qbuf + ((size_t)b * NH  + h)   * SEQ * HDIM;
  const bf16_t* Kg = kbuf + ((size_t)b * NKV + hkv) * SEQ * HDIM;
  const bf16_t* Vg = vbuf + ((size_t)b * NKV + hkv) * SEQ * HDIM;

  __shared__ alignas(64) bf16_t ldsK[2][32 * HDIM]; // [buf][kv][hd], TDM-filled
  __shared__ alignas(64) bf16_t ldsVt[HDIM * 32];   // [hd][kv] (transposed)
  __shared__ alignas(64) bf16_t ldsP[4][16 * 32];   // per-wave P staging

  // Q A-fragments, 4 K-chunks of 32 covering HD=128; loaded once.
  v16bf qf[4];
#pragma unroll
  for (int c = 0; c < 4; ++c)
    qf[c] = load_afrag(Q + (size_t)(qbase + col) * HDIM + c * 32, khalf);

  v8f acc[8];
  float rmax[8], rsum[8];
#pragma unroll
  for (int ni = 0; ni < 8; ++ni) acc[ni] = zero_v8f();
#pragma unroll
  for (int j = 0; j < 8; ++j) { rmax[j] = -3.0e38f; rsum[j] = 0.0f; }

  const float scale = 0.08838834764831845f;  // 1/sqrt(128)
  const int ntiles = (qbase_wg + 64) / 32;   // causal upper bound for this wg

  // Prologue: DMA K tile 0 into buffer 0 (single issuing wave; TDM ignores EXEC).
  if (wid == 0) {
    tdm_load_2d_bf16(Kg, lds_off(&ldsK[0][0]), 32, HDIM, SEQ);
  }

  for (int t = 0; t < ntiles; ++t) {
    const int kv0 = t * 32;
    __syncthreads();  // previous tile's compute done; ldsVt reusable

    // V tile 32x128 -> ldsVt transposed (vector path, overlaps TDM).
#pragma unroll
    for (int c = 0; c < 2; ++c) {
      int chunk = tid * 2 + c;
      int row = chunk >> 3;   // kv row 0..31
      int c16 = chunk & 7;    // 16-elem chunk along hd
      v16bf vv = *(const v16bf*)(Vg + (size_t)(kv0 + row) * HDIM + c16 * 16);
#pragma unroll
      for (int i = 0; i < 16; ++i) ldsVt[(c16 * 16 + i) * 32 + row] = vv[i];
    }

    // Pipeline: issue DMA for tile t+1, then drain to <=1 outstanding so
    // tile t's buffer is complete before the publish barrier.
    if (wid == 0) {
      if (t + 1 < ntiles) {
        tdm_load_2d_bf16(Kg + (size_t)(kv0 + 32) * HDIM,
                         lds_off(&ldsK[(t + 1) & 1][0]),
                         32, HDIM, (unsigned)(SEQ - (kv0 + 32)));
        __builtin_amdgcn_s_wait_tensorcnt(1);
      } else {
        __builtin_amdgcn_s_wait_tensorcnt(0);
      }
    }
    __syncthreads();

    const bf16_t* Kt = &ldsK[t & 1][0];

    // S = Q K^T : two 16x16 subtiles, inner dim 128 via 4 chained WMMAs.
    v8f s0 = zero_v8f(), s1 = zero_v8f();
#pragma unroll
    for (int c = 0; c < 4; ++c) {
      v16bf kb0 = *(const v16bf*)&Kt[(col)      * HDIM + c * 32 + khalf * 16];
      v16bf kb1 = *(const v16bf*)&Kt[(16 + col) * HDIM + c * 32 + khalf * 16];
      s0 = wmma_bf16(qf[c], kb0, s0);
      s1 = wmma_bf16(qf[c], kb1, s1);
    }

    // Online softmax per q-row (16-lane group reductions; wave32).
#pragma unroll
    for (int j = 0; j < 8; ++j) {
      int r  = j + 8 * khalf;
      int qg = qbase + r;
      int kg0 = kv0 + col, kg1 = kv0 + 16 + col;
      float e0 = (kg0 <= qg) ? s0[j] * scale : -3.0e38f;
      float e1 = (kg1 <= qg) ? s1[j] * scale : -3.0e38f;
      float mx = fmaxf(e0, e1);
      mx = fmaxf(mx, __shfl_xor(mx, 1, 32));
      mx = fmaxf(mx, __shfl_xor(mx, 2, 32));
      mx = fmaxf(mx, __shfl_xor(mx, 4, 32));
      mx = fmaxf(mx, __shfl_xor(mx, 8, 32));
      float mnew = fmaxf(rmax[j], mx);
      float corr = __expf(rmax[j] - mnew);
      rmax[j] = mnew;
      float p0 = __expf(e0 - mnew);
      float p1 = __expf(e1 - mnew);
      float rs = p0 + p1;
      rs += __shfl_xor(rs, 1, 32);
      rs += __shfl_xor(rs, 2, 32);
      rs += __shfl_xor(rs, 4, 32);
      rs += __shfl_xor(rs, 8, 32);
      rsum[j] = rsum[j] * corr + rs;
#pragma unroll
      for (int ni = 0; ni < 8; ++ni) acc[ni][j] *= corr;
      // Stage P (C-layout -> LDS) for re-read in A-fragment layout.
      ldsP[wid][r * 32 + col]      = (bf16_t)p0;
      ldsP[wid][r * 32 + 16 + col] = (bf16_t)p1;
    }

    // O += P(16x32) @ V(32x128): 8 WMMAs across hd.
    v16bf pf = load_afrag(&ldsP[wid][col * 32], khalf);
#pragma unroll
    for (int ni = 0; ni < 8; ++ni) {
      v16bf vf = *(const v16bf*)&ldsVt[(ni * 16 + col) * 32 + khalf * 16];
      acc[ni] = wmma_bf16(pf, vf, acc[ni]);
    }
  }

  // Normalize and store to [B, S, NH, HD].
#pragma unroll
  for (int j = 0; j < 8; ++j) {
    int r  = j + 8 * khalf;
    int sq = qbase + r;
    float inv = 1.0f / rsum[j];
#pragma unroll
    for (int ni = 0; ni < 8; ++ni) {
      int hd = ni * 16 + col;
      abuf[(((size_t)b * SEQ + sq) * NH + h) * HDIM + hd] =
          (bf16_t)(acc[ni][j] * inv);
    }
  }
}

// ---------------------------------------------------------------------------
extern "C" void kernel_launch(void* const* d_in, const int* in_sizes, int n_in,
                              void* d_out, int out_size, void* d_ws, size_t ws_size,
                              hipStream_t stream)
{
  (void)in_sizes; (void)n_in; (void)out_size; (void)ws_size;
  const float* x  = (const float*)d_in[0];
  // d_in[1] is the causal mask; causality is applied analytically.
  const float* Wq = (const float*)d_in[2];
  const float* Wk = (const float*)d_in[3];
  const float* Wv = (const float*)d_in[4];
  const float* Wo = (const float*)d_in[5];

  // Workspace: q 16MB | k 2MB | v 2MB | attn_out 16MB  (36MB total, bf16)
  char* ws = (char*)d_ws;
  const size_t qbytes  = (size_t)B_ * NH  * SEQ * HDIM * sizeof(bf16_t);
  const size_t kvbytes = (size_t)B_ * NKV * SEQ * HDIM * sizeof(bf16_t);
  bf16_t* qbuf = (bf16_t*)ws;
  bf16_t* kbuf = (bf16_t*)(ws + qbytes);
  bf16_t* vbuf = (bf16_t*)(ws + qbytes + kvbytes);
  bf16_t* abuf = (bf16_t*)(ws + qbytes + 2 * kvbytes);

  const int M = B_ * SEQ;  // 4096
  dim3 blk(256);

  // QKV projections (f32 inputs -> bf16 outputs in [B,Hx,S,HD] layout).
  gemm_wmma_kernel<0><<<dim3((NH * HDIM) / 128, M / 128), blk, 0, stream>>>(
      x, Wq, qbuf, M, NH * HDIM, DMODEL, 1, SEQ, NH);
  gemm_wmma_kernel<0><<<dim3((NKV * HDIM) / 128, M / 128), blk, 0, stream>>>(
      x, Wk, kbuf, M, NKV * HDIM, DMODEL, 1, SEQ, NKV);
  gemm_wmma_kernel<0><<<dim3((NKV * HDIM) / 128, M / 128), blk, 0, stream>>>(
      x, Wv, vbuf, M, NKV * HDIM, DMODEL, 1, SEQ, NKV);

  // RoPE on q and k (in place).
  size_t tq = (size_t)B_ * NH * SEQ * (HDIM / 2);
  rope_kernel<<<(unsigned)((tq + 255) / 256), 256, 0, stream>>>(qbuf, NH);
  size_t tk = (size_t)B_ * NKV * SEQ * (HDIM / 2);
  rope_kernel<<<(unsigned)((tk + 255) / 256), 256, 0, stream>>>(kbuf, NKV);

  // Causal flash attention with GQA sharing + TDM-pipelined K tiles.
  attn_kernel<<<B_ * NH * (SEQ / 64), 128, 0, stream>>>(qbuf, kbuf, vbuf, abuf);

  // Output projection: bf16 A path, f32 output.
  gemm_wmma_kernel<1><<<dim3(DMODEL / 128, M / 128), blk, 0, stream>>>(
      abuf, Wo, d_out, M, DMODEL, NH * HDIM, 0, SEQ, NH);
}
